// SpanDynamicConvAttention_19524921328247
// MI455X (gfx1250) — matrix operand order repaired and checked by reading, not compiled, not run on hardware
//
#include <hip/hip_runtime.h>
#include <hip/hip_bf16.h>

typedef __attribute__((ext_vector_type(16))) __bf16       v16bf;
typedef __attribute__((ext_vector_type(8)))  float        v8f;
typedef __attribute__((ext_vector_type(8)))  unsigned int v8u;

#define S_LEN  1024
#define D_DIM  384
#define HALFD  192     // D_DIM/2 (element pairs)
#define H_NUM  6
#define K_SZ   9
#define NK     54      // K*H rows of Wk
#define OUT_N  384
#define MTILE  32      // seq rows per workgroup
#define SW     392     // padded LDS stride in bf16 elements (384 + 8)

// round-to-nearest-even f32 -> bf16 (raw ushort)
static __device__ __forceinline__ unsigned short f2bfu(float f) {
  union { float f; unsigned u; } x; x.f = f;
  unsigned r = x.u + 0x7FFFu + ((x.u >> 16) & 1u);
  return (unsigned short)(r >> 16);
}
static __device__ __forceinline__ float bf2f(unsigned short h) {
  union { unsigned u; float f; } x; x.u = ((unsigned)h) << 16;
  return x.f;
}
static __device__ __forceinline__ unsigned packbf(float a, float b) {
  return (unsigned)f2bfu(a) | ((unsigned)f2bfu(b) << 16);
}

// A fragment (16x32 bf16) for one lane from a row-major bf16 LDS tile.
// base points at &tile[row*SW + k0 + 8*half]; dwords {0..3} then ushort+16 {0..3}.
static __device__ __forceinline__ v16bf load_a_lds(const unsigned short* base) {
  const uint4* p = (const uint4*)base;
  uint4 lo = p[0];   // K = 8h + {0,2,4,6} (+k0)
  uint4 hi = p[2];   // K = 8h + {16,18,20,22}
  v8u au;
  au[0] = lo.x; au[1] = lo.y; au[2] = lo.z; au[3] = lo.w;
  au[4] = hi.x; au[5] = hi.y; au[6] = hi.z; au[7] = hi.w;
  return __builtin_bit_cast(v16bf, au);
}

// B fragment (32x16 bf16) for one lane: f32 at offsets 0..7 and 16..23 of one weight row,
// packed to bf16 pairs; mask==0 zeroes the column (padding rows).
static __device__ __forceinline__ v16bf load_b_f32(const float* rowp, unsigned mask) {
  const float4* p = (const float4*)rowp;
  float4 c0 = p[0], c1 = p[1], c2 = p[4], c3 = p[5];
  v8u bu;
  bu[0] = packbf(c0.x, c0.y) & mask; bu[1] = packbf(c0.z, c0.w) & mask;
  bu[2] = packbf(c1.x, c1.y) & mask; bu[3] = packbf(c1.z, c1.w) & mask;
  bu[4] = packbf(c2.x, c2.y) & mask; bu[5] = packbf(c2.z, c2.w) & mask;
  bu[6] = packbf(c3.x, c3.y) & mask; bu[7] = packbf(c3.z, c3.w) & mask;
  return __builtin_bit_cast(v16bf, bu);
}

__global__ __launch_bounds__(256)
void sdca_fused_kernel(const float* __restrict__ q,  const float* __restrict__ ks,
                       const float* __restrict__ v,  const float* __restrict__ Wk,
                       const float* __restrict__ bk, const float* __restrict__ Wp,
                       const float* __restrict__ bp, float* __restrict__ out) {
  __shared__ unsigned short sBuf[40 * SW];       // t-staging, then v window (bf16)
  __shared__ float          sLog[MTILE * 64];    // logits -> dk (in place)
  __shared__ unsigned short sOut[MTILE * SW];    // conv result (bf16, GEMM2 A)

  const int tid  = threadIdx.x;
  const int lane = tid & 31;
  const int wave = tid >> 5;
  const int half = lane >> 4;     // 0: lanes 0-15, 1: lanes 16-31
  const int l16  = lane & 15;

  const int b  = blockIdx.x >> 5;             // 32 s-tiles per batch
  const int s0 = (blockIdx.x & 31) * MTILE;

  // ---------------- P0: t = q*ks tile -> bf16 LDS (GEMM1 A) ----------------
  for (int i = tid; i < MTILE * HALFD; i += 256) {
    int r = i / HALFD, d2 = (i - r * HALFD) * 2;
    size_t g = ((size_t)b * S_LEN + (size_t)(s0 + r)) * D_DIM + d2;
    float2 qa = *(const float2*)(q + g);
    float2 ka = *(const float2*)(ks + g);
    *(unsigned*)&sBuf[r * SW + d2] = packbf(qa.x * ka.x, qa.y * ka.y);
  }
  __syncthreads();

  // ---------------- P1: logits = t . Wk^T via bf16 WMMA --------------------
  {
    int mt = wave & 1;                 // 2 m-tiles of 16 rows
    int nt = wave >> 1;                // 4 n-tiles (54 -> padded 64)
    int nrow = nt * 16 + l16;          // Wk row feeding this lane's B column
    unsigned mask = (nrow < NK) ? 0xFFFFFFFFu : 0u;
    const float* wkrow = Wk + (size_t)(nrow < NK ? nrow : NK - 1) * D_DIM + 8 * half;
    const unsigned short* abase = &sBuf[(mt * 16 + l16) * SW + 8 * half];
    v8f acc = {};
#pragma unroll 2
    for (int kk = 0; kk < 12; ++kk) {
      int k0 = kk * 32;
      v16bf am = load_a_lds(abase + k0);
      v16bf bm = load_b_f32(wkrow + k0, mask);
      acc = __builtin_amdgcn_wmma_f32_16x16x32_bf16(
          false, am, false, bm, (short)0, acc, false, false);
    }
    float bkv = (nrow < NK) ? bk[nrow] : 0.0f;
#pragma unroll
    for (int r = 0; r < 8; ++r) {
      int row = mt * 16 + r + 8 * half;            // C/D layout: M = r + 8*half
      sLog[row * 64 + nt * 16 + l16] = acc[r] + bkv;
    }
  }
  __syncthreads();

  // ------- P2: softmax over 9 taps (in place); P3: stage v window ----------
  if (tid < MTILE * H_NUM) {
    int row = tid / H_NUM, h = tid - row * H_NUM;
    float* p = &sLog[row * 64 + h * K_SZ];
    float m = p[0];
#pragma unroll
    for (int k = 1; k < K_SZ; ++k) m = fmaxf(m, p[k]);
    float e[K_SZ], s = 0.f;
#pragma unroll
    for (int k = 0; k < K_SZ; ++k) { e[k] = __expf(p[k] - m); s += e[k]; }
    float inv = 1.0f / s;
#pragma unroll
    for (int k = 0; k < K_SZ; ++k) p[k] = e[k] * inv;
  }
  for (int i = tid; i < 40 * HALFD; i += 256) {    // rows s0-4 .. s0+35
    int r = i / HALFD, d2 = (i - r * HALFD) * 2;
    int g = s0 + r - 4;
    unsigned w = 0u;
    if (g >= 0 && g < S_LEN) {
      float2 va = *(const float2*)(v + ((size_t)b * S_LEN + g) * D_DIM + d2);
      w = packbf(va.x, va.y);
    }
    *(unsigned*)&sBuf[r * SW + d2] = w;
  }
  __syncthreads();

  // ---------------- P4: dynamic conv -> sOut bf16 (GEMM2 A) ----------------
  for (int i = tid; i < MTILE * HALFD; i += 256) {
    int r = i / HALFD, d2 = (i - r * HALFD) * 2;
    int h = d2 >> 6;                               // head = d / 64 (pair-safe)
    const float* wk9 = &sLog[r * 64 + h * K_SZ];
    float a0 = 0.f, a1 = 0.f;
#pragma unroll
    for (int k = 0; k < K_SZ; ++k) {
      unsigned w = *(const unsigned*)&sBuf[(r + k) * SW + d2];
      float dk = wk9[k];
      a0 += bf2f((unsigned short)w) * dk;
      a1 += bf2f((unsigned short)(w >> 16)) * dk;
    }
    *(unsigned*)&sOut[r * SW + d2] = packbf(a0, a1);
  }
  __syncthreads();

  // ---------------- P5: out = conv . Wp^T + bp via bf16 WMMA ---------------
  for (int nt = wave; nt < 24; nt += 8) {          // 24 n-tiles over OUT=384
    int nrow = nt * 16 + l16;                      // Wp row for this column
    const float* wprow = Wp + (size_t)nrow * D_DIM + 8 * half;
    const unsigned short* abase0 = &sBuf[0];       // silence unused warnings path
    (void)abase0;
    const unsigned short* a0base = &sOut[l16 * SW + 8 * half];
    const unsigned short* a1base = &sOut[(16 + l16) * SW + 8 * half];
    v8f acc0 = {}, acc1 = {};
#pragma unroll 4
    for (int kk = 0; kk < 12; ++kk) {
      int k0 = kk * 32;
      __builtin_prefetch(wprow + k0 + 32, 0, 1);   // next k-slice of Wp (L2 hit)
      v16bf bm = load_b_f32(wprow + k0, 0xFFFFFFFFu);
      v16bf am0 = load_a_lds(a0base + k0);
      v16bf am1 = load_a_lds(a1base + k0);
      acc0 = __builtin_amdgcn_wmma_f32_16x16x32_bf16(
          false, am0, false, bm, (short)0, acc0, false, false);
      acc1 = __builtin_amdgcn_wmma_f32_16x16x32_bf16(
          false, am1, false, bm, (short)0, acc1, false, false);
    }
    float bpv = bp[nrow];
#pragma unroll
    for (int r = 0; r < 8; ++r) {
      int srow = s0 + r + 8 * half;
      out[((size_t)b * S_LEN + srow) * OUT_N + nrow]      = acc0[r] + bpv;
      out[((size_t)b * S_LEN + srow + 16) * OUT_N + nrow] = acc1[r] + bpv;
    }
  }
}

extern "C" void kernel_launch(void* const* d_in, const int* in_sizes, int n_in,
                              void* d_out, int out_size, void* d_ws, size_t ws_size,
                              hipStream_t stream) {
  (void)in_sizes; (void)n_in; (void)out_size; (void)d_ws; (void)ws_size;
  const float* q  = (const float*)d_in[0];
  const float* ks = (const float*)d_in[1];
  const float* v  = (const float*)d_in[2];
  const float* Wk = (const float*)d_in[3];
  const float* bk = (const float*)d_in[4];
  const float* Wp = (const float*)d_in[5];
  const float* bp = (const float*)d_in[6];
  float* out = (float*)d_out;

  dim3 grid(16 * (S_LEN / MTILE));   // 512 workgroups
  dim3 block(256);                   // 8 wave32 waves
  sdca_fused_kernel<<<grid, block, 0, stream>>>(q, ks, v, Wk, bk, Wp, bp, out);
}
